// EGNNLayer_4080218931280
// MI455X (gfx1250) — matrix operand order repaired and checked by reading, not compiled, not run on hardware
//
#include <hip/hip_runtime.h>

typedef __attribute__((ext_vector_type(4)))  _Float16 v4h;
typedef __attribute__((ext_vector_type(8)))  _Float16 v8h;
typedef __attribute__((ext_vector_type(16))) _Float16 v16h;
typedef __attribute__((ext_vector_type(4)))  float    v4f;
typedef __attribute__((ext_vector_type(8)))  float    v8f;

#define EPS_F    1e-8f
#define LN_EPS_F 1e-5f

__device__ __forceinline__ float fast_rcp(float x) {
#if __has_builtin(__builtin_amdgcn_rcpf)
    return __builtin_amdgcn_rcpf(x);
#else
    return 1.0f / x;
#endif
}
__device__ __forceinline__ float silu_f(float v) {
    // v * sigmoid(v) with hardware v_rcp_f32 (f16-GEMM pipeline; approx rcp is fine)
    return v * fast_rcp(1.0f + __expf(-v));
}
__device__ __forceinline__ float tanh_f(float s) {
    // tanh(s) = 1 - 2/(e^{2s}+1); exp overflow -> +1, underflow -> -1 (correct limits)
    return 1.0f - 2.0f * fast_rcp(__expf(2.0f * s) + 1.0f);
}

// ---------------------------------------------------------------------------
// Dual 16x16 WMMA tiles sharing one A fragment: one wave computes output
// columns [nt0*16..) and [nt1*16..) for M-tile mt. A is row-major f16 in LDS
// (ASTRIDE halfs/row); B is fragment-packed (chunks of (kc,nt), 512 halfs,
// lane-major, per pack_w_kernel).
// A fragment layout used here (contiguous k, same order as B):
//   lane<16 : M=lane,    K = 0..15 (one contiguous 16-half run)
//   lane>=16: M=lane-16, K = 16..31
// B fragment: lane L holds column N=(L&15), K = (L>>4)*16 + i, i=0..15.
// ---------------------------------------------------------------------------
template<int KC, int ASTRIDE>
__device__ __forceinline__ void wmma_dual16(const _Float16* A, const _Float16* Bpk,
                                            int mt, int nt0, int nt1, int lane,
                                            float bias0, float bias1,
                                            v8f& acc0, v8f& acc1) {
    const int l15 = lane & 15;
    const int sel = lane >> 4;
    acc0 = (v8f){bias0, bias0, bias0, bias0, bias0, bias0, bias0, bias0};
    acc1 = (v8f){bias1, bias1, bias1, bias1, bias1, bias1, bias1, bias1};
#pragma unroll
    for (int kc = 0; kc < KC; ++kc) {
        const _Float16* ar = A + (mt * 16 + l15) * ASTRIDE + kc * 32 + sel * 16;
        v8h lo = *(const v8h*)ar;            // K = kc*32 + sel*16 .. +7
        v8h hi = *(const v8h*)(ar + 8);      // K = kc*32 + sel*16 + 8 .. +15
        v16h a = __builtin_shufflevector(lo, hi, 0,1,2,3,4,5,6,7,8,9,10,11,12,13,14,15);
        v16h b0 = *(const v16h*)(Bpk + (kc * 4 + nt0) * 512 + lane * 16);
        v16h b1 = *(const v16h*)(Bpk + (kc * 4 + nt1) * 512 + lane * 16);
        acc0 = __builtin_amdgcn_wmma_f32_16x16x32_f16(false, a, false, b0, (short)0, acc0, false, false);
        acc1 = __builtin_amdgcn_wmma_f32_16x16x32_f16(false, a, false, b1, (short)0, acc1, false, false);
    }
}

// ---------------------------------------------------------------------------
__global__ void zero2_kernel(float* __restrict__ a, long long na,
                             float* __restrict__ b, long long nb) {
    long long i = (long long)blockIdx.x * blockDim.x + threadIdx.x;
    long long s = (long long)gridDim.x * blockDim.x;
    for (long long k = i; k < na; k += s) a[k] = 0.0f;
    for (long long k = i; k < nb; k += s) b[k] = 0.0f;
}

// Repack a K x 64 f32 weight into WMMA B fragments (f16), K padded to Kpad.
__global__ void pack_w_kernel(const float* __restrict__ src, _Float16* __restrict__ dst,
                              int K, int Kpad) {
    int idx = blockIdx.x * blockDim.x + threadIdx.x;
    int total = (Kpad / 32) * 4 * 512;
    if (idx >= total) return;
    int chunk  = idx >> 9;        // (kc*4 + nt)
    int within = idx & 511;
    int lane   = within >> 4;
    int i      = within & 15;
    int kc = chunk >> 2;
    int nt = chunk & 3;
    int k = kc * 32 + (lane >> 4) * 16 + i;
    int n = nt * 16 + (lane & 15);
    float v = (k < K) ? src[k * 64 + n] : 0.0f;
    dst[idx] = (_Float16)v;
}

// ---------------------------------------------------------------------------
// Edge kernel: 64 edges per block, 256 threads (8 waves).
// message MLP + coord gate, all GEMMs via WMMA; scatter with f32 atomics.
// Wave wv owns M-tile (wv&3) and N-tiles {2*(wv>>2), 2*(wv>>2)+1}.
// ---------------------------------------------------------------------------
__launch_bounds__(256)
__global__ void edge_kernel(const float* __restrict__ h, const float* __restrict__ x,
                            const int* __restrict__ eidx, const float* __restrict__ eattr,
                            const _Float16* __restrict__ w1pk, const float* __restrict__ b1,
                            const _Float16* __restrict__ w2pk, const float* __restrict__ b2,
                            const _Float16* __restrict__ cw1pk, const float* __restrict__ cb1,
                            const float* __restrict__ cw2,
                            float* __restrict__ mi_acc,   // [N,64] accumulator (d_out h region)
                            float* __restrict__ tr_acc,   // [N,3]  accumulator (d_out x region)
                            float* __restrict__ cnt,      // [N]
                            int E) {
    __shared__ _Float16 sIn[64][160];   // edge_input, padded 145 -> 160
    __shared__ _Float16 sH1[64][64];    // silu(hidden1)
    __shared__ _Float16 sM [64][64];    // m_ij (f16 copy for GEMM3)
    __shared__ float    sGT[64][64];    // TRANSPOSED silu(m_ij@cw1+cb1): sGT[n][edge]
    __shared__ int      sRow[64], sCol[64];
    __shared__ float    sLd[64];
    __shared__ float    sRad[64][3];

    const int t = threadIdx.x;
    const long long e0 = (long long)blockIdx.x * 64;

    if (t < 64) {
        long long e = e0 + t;
        int r = -1, c = 0;
        float ld = 0.f, r0 = 0.f, r1 = 0.f, r2 = 0.f;
        if (e < (long long)E) {
            r = eidx[e];
            c = eidx[(long long)E + e];
            float d0 = x[r * 3 + 0] - x[c * 3 + 0];
            float d1 = x[r * 3 + 1] - x[c * 3 + 1];
            float d2 = x[r * 3 + 2] - x[c * 3 + 2];
            float ds = d0 * d0 + d1 * d1 + d2 * d2;
            float dist = sqrtf(ds + EPS_F);
            float inv = fast_rcp(dist + EPS_F);
            ld = log1pf(ds);
            r0 = d0 * inv; r1 = d1 * inv; r2 = d2 * inv;
        }
        sRow[t] = r; sCol[t] = c; sLd[t] = ld;
        sRad[t][0] = r0; sRad[t][1] = r1; sRad[t][2] = r2;
    }
    __syncthreads();

    // Stage [h[row] | h[col]] (j<128) with float4 loads -> b128 VMEM.
    for (int idx = t; idx < 64 * 32; idx += 256) {
        int e = idx >> 5;            // edge in block
        int q = idx & 31;            // float4 chunk, j = q*4
        int r = sRow[e];
        v4f v = {0.f, 0.f, 0.f, 0.f};
        if (r >= 0) {
            const float* src = (q < 16) ? &h[(long long)r * 64 + q * 4]
                                        : &h[(long long)sCol[e] * 64 + (q - 16) * 4];
            v = *(const v4f*)src;
        }
        v4h hv = {(_Float16)v[0], (_Float16)v[1], (_Float16)v[2], (_Float16)v[3]};
        *(v4h*)&sIn[e][q * 4] = hv;
    }
    // Tail: [log1p | edge_attr | zero-pad] (j = 128..159)
    for (int idx = t; idx < 64 * 32; idx += 256) {
        int e = idx >> 5;
        int j = 128 + (idx & 31);
        float v = 0.f;
        int r = sRow[e];
        if (r >= 0) {
            if (j == 128)     v = sLd[e];
            else if (j < 145) v = eattr[(e0 + e) * 16 + (j - 129)];
        }
        sIn[e][j] = (_Float16)v;
    }
    __syncthreads();

    const int wv = t >> 5, lane = t & 31;
    const int l15 = lane & 15, sel = lane >> 4;
    const int mt  = wv & 3;
    const int nt0 = (wv >> 2) * 2;
    const int nt1 = nt0 + 1;
    const int n0g = nt0 * 16 + l15;     // global output column for acc0
    const int n1g = nt1 * 16 + l15;     // global output column for acc1

    v8f a0, a1;

    // GEMM1: [64x160] @ [160x64] + b1, silu
    wmma_dual16<5, 160>(&sIn[0][0], w1pk, mt, nt0, nt1, lane, b1[n0g], b1[n1g], a0, a1);
#pragma unroll
    for (int v = 0; v < 8; ++v) {
        int m = mt * 16 + v + sel * 8;               // C layout: M = v (+8 for hi lanes)
        sH1[m][n0g] = (_Float16)silu_f(a0[v]);
        sH1[m][n1g] = (_Float16)silu_f(a1[v]);
    }
    __syncthreads();

    // GEMM2: m_ij = silu(h1 @ w2 + b2); scatter into mi_acc
    wmma_dual16<2, 64>(&sH1[0][0], w2pk, mt, nt0, nt1, lane, b2[n0g], b2[n1g], a0, a1);
#pragma unroll
    for (int v = 0; v < 8; ++v) {
        int m = mt * 16 + v + sel * 8;
        float f0 = silu_f(a0[v]);
        float f1 = silu_f(a1[v]);
        sM[m][n0g] = (_Float16)f0;
        sM[m][n1g] = (_Float16)f1;
        int r = sRow[m];
        if (r >= 0) {
            atomicAdd(&mi_acc[(long long)r * 64 + n0g], f0);
            atomicAdd(&mi_acc[(long long)r * 64 + n1g], f1);
        }
    }
    __syncthreads();

    // GEMM3: gate hidden = silu(m_ij @ cw1 + cb1), stored transposed
    wmma_dual16<2, 64>(&sM[0][0], cw1pk, mt, nt0, nt1, lane, cb1[n0g], cb1[n1g], a0, a1);
#pragma unroll
    for (int v = 0; v < 8; ++v) {
        int m = mt * 16 + v + sel * 8;
        sGT[n0g][m] = silu_f(a0[v]);                 // transposed: [feature][edge]
        sGT[n1g][m] = silu_f(a1[v]);
    }
    __syncthreads();

    // gate = tanh(g @ cw2); trans = radial * gate * 0.1; scatter.
    // 4 threads per edge, 16-element partials, wave32 shfl reduction.
    {
        int e = t >> 2;           // edge in block
        int q = t & 3;            // quarter
        float s = 0.f;
#pragma unroll
        for (int jj = 0; jj < 16; ++jj) {
            int j = q * 16 + jj;
            s += sGT[j][e] * cw2[j];
        }
        s += __shfl_xor(s, 1, 32);
        s += __shfl_xor(s, 2, 32);
        if (q == 0) {
            int r = sRow[e];
            if (r >= 0) {
                float g = 0.1f * tanh_f(s);
                atomicAdd(&tr_acc[(long long)r * 3 + 0], sRad[e][0] * g);
                atomicAdd(&tr_acc[(long long)r * 3 + 1], sRad[e][1] * g);
                atomicAdd(&tr_acc[(long long)r * 3 + 2], sRad[e][2] * g);
                atomicAdd(&cnt[r], 1.0f);
            }
        }
    }
}

// ---------------------------------------------------------------------------
// Node kernel: 64 nodes per block, 256 threads. Node MLP + residual + LN + x.
// mi_acc / tr_acc are read and then overwritten in-place (block-local rows).
// ---------------------------------------------------------------------------
__launch_bounds__(256)
__global__ void node_kernel(const float* __restrict__ h, const float* __restrict__ x,
                            const _Float16* __restrict__ nw1pk, const float* __restrict__ nb1,
                            const _Float16* __restrict__ nw2pk, const float* __restrict__ nb2,
                            const float* __restrict__ ln_g, const float* __restrict__ ln_b,
                            const float* __restrict__ cnt,
                            float* __restrict__ hout_mi,  // in: m_i sums, out: h_out [N,64]
                            float* __restrict__ xout_tr,  // in: trans sums, out: x_new [N,3]
                            int N) {
    __shared__ _Float16 sIn[64][128];
    __shared__ _Float16 sH1[64][64];
    __shared__ float    sHNT[64][64];   // TRANSPOSED mlp output: sHNT[n][node]
    __shared__ float    sInv[64];

    const int t = threadIdx.x;
    const long long n0 = (long long)blockIdx.x * 64;

    if (t < 64) {
        long long nd = n0 + t;
        sInv[t] = (nd < (long long)N) ? fast_rcp(cnt[nd] + EPS_F) : 0.0f;
    }
    __syncthreads();

    // Stage [h | m_i] as f16 with float4 loads
    for (int idx = t; idx < 64 * 32; idx += 256) {
        int e = idx >> 5;
        int q = idx & 31;
        long long nd = n0 + e;
        v4f v = {0.f, 0.f, 0.f, 0.f};
        if (nd < (long long)N) {
            if (q < 16) {
                v = *(const v4f*)&h[nd * 64 + q * 4];
            } else {
                v = *(const v4f*)&hout_mi[nd * 64 + (q - 16) * 4];
                v *= sInv[e];               // m_i = sum/(cnt+eps)
            }
        }
        v4h hv = {(_Float16)v[0], (_Float16)v[1], (_Float16)v[2], (_Float16)v[3]};
        *(v4h*)&sIn[e][q * 4] = hv;
    }
    __syncthreads();

    const int wv = t >> 5, lane = t & 31;
    const int l15 = lane & 15, sel = lane >> 4;
    const int mt  = wv & 3;
    const int nt0 = (wv >> 2) * 2;
    const int nt1 = nt0 + 1;
    const int n0g = nt0 * 16 + l15;
    const int n1g = nt1 * 16 + l15;

    v8f a0, a1;

    // GEMM1: [64x128] @ [128x64] + b1, silu
    wmma_dual16<4, 128>(&sIn[0][0], nw1pk, mt, nt0, nt1, lane, nb1[n0g], nb1[n1g], a0, a1);
#pragma unroll
    for (int v = 0; v < 8; ++v) {
        int m = mt * 16 + v + sel * 8;
        sH1[m][n0g] = (_Float16)silu_f(a0[v]);
        sH1[m][n1g] = (_Float16)silu_f(a1[v]);
    }
    __syncthreads();

    // GEMM2: mlp_out = h1 @ w2 + b2, stored transposed (residual added in LN stage)
    wmma_dual16<2, 64>(&sH1[0][0], nw2pk, mt, nt0, nt1, lane, nb2[n0g], nb2[n1g], a0, a1);
#pragma unroll
    for (int v = 0; v < 8; ++v) {
        int m = mt * 16 + v + sel * 8;
        sHNT[n0g][m] = a0[v];               // transposed: [feature][node]
        sHNT[n1g][m] = a1[v];
    }
    __syncthreads();

    // Residual + LayerNorm + x update. sHNT[j][t] reads are conflict-free.
    if (t < 64) {
        long long nd = n0 + t;
        if (nd < (long long)N) {
            // pass 1: add residual (float4 row loads of h), accumulate mean
            float mu = 0.f;
#pragma unroll
            for (int q = 0; q < 16; ++q) {
                v4f hv = *(const v4f*)&h[nd * 64 + q * 4];
#pragma unroll
                for (int k = 0; k < 4; ++k) {
                    float hn = sHNT[q * 4 + k][t] + hv[k];
                    sHNT[q * 4 + k][t] = hn;
                    mu += hn;
                }
            }
            mu *= (1.0f / 64.0f);
            // pass 2: variance
            float var = 0.f;
#pragma unroll
            for (int j = 0; j < 64; ++j) { float d = sHNT[j][t] - mu; var += d * d; }
            var *= (1.0f / 64.0f);
            float rstd = rsqrtf(var + LN_EPS_F);
            // pass 3: normalize, scale/shift, store
#pragma unroll
            for (int j = 0; j < 64; ++j)
                hout_mi[nd * 64 + j] = (sHNT[j][t] - mu) * rstd * ln_g[j] + ln_b[j];
            // x_new = x + trans/(cnt+eps)
            float iv = sInv[t];
#pragma unroll
            for (int k = 0; k < 3; ++k)
                xout_tr[nd * 3 + k] = x[nd * 3 + k] + xout_tr[nd * 3 + k] * iv;
        }
    }
}

// ---------------------------------------------------------------------------
extern "C" void kernel_launch(void* const* d_in, const int* in_sizes, int n_in,
                              void* d_out, int out_size, void* d_ws, size_t ws_size,
                              hipStream_t stream) {
    (void)n_in; (void)out_size; (void)ws_size;

    const float* h     = (const float*)d_in[0];
    const float* x     = (const float*)d_in[1];
    const int*   eidx  = (const int*)d_in[2];
    const float* eattr = (const float*)d_in[3];
    const float* msw1  = (const float*)d_in[4];
    const float* msb1  = (const float*)d_in[5];
    const float* msw2  = (const float*)d_in[6];
    const float* msb2  = (const float*)d_in[7];
    const float* nw1   = (const float*)d_in[8];
    const float* nb1   = (const float*)d_in[9];
    const float* nw2   = (const float*)d_in[10];
    const float* nb2   = (const float*)d_in[11];
    const float* cw1   = (const float*)d_in[12];
    const float* cb1   = (const float*)d_in[13];
    const float* cw2   = (const float*)d_in[14];
    const float* lng   = (const float*)d_in[15];
    const float* lnb   = (const float*)d_in[16];

    const int N = in_sizes[1] / 3;    // x is [N,3]
    const int E = in_sizes[3] / 16;   // edge_attr is [E,16]

    float* out    = (float*)d_out;
    float* mi_acc = out;                          // [N,64] — m_i sums, then h_out
    float* tr_acc = out + (long long)N * 64;      // [N,3]  — trans sums, then x_new

    // workspace: packed f16 weights, then counts
    char* ws = (char*)d_ws;
    _Float16* w1pk  = (_Float16*)(ws);                       // 160x64 -> 10240 halfs
    _Float16* w2pk  = (_Float16*)(ws + 20480);               //  64x64 ->  4096
    _Float16* cw1pk = (_Float16*)(ws + 28672);               //  64x64 ->  4096
    _Float16* nw1pk = (_Float16*)(ws + 36864);               // 128x64 ->  8192
    _Float16* nw2pk = (_Float16*)(ws + 53248);               //  64x64 ->  4096
    float*    cnt   = (float*)   (ws + 61440);               // [N]

    zero2_kernel<<<2048, 256, 0, stream>>>(out, (long long)N * 67, cnt, (long long)N);

    pack_w_kernel<<<(10240 + 255) / 256, 256, 0, stream>>>(msw1, w1pk, 145, 160);
    pack_w_kernel<<<(4096 + 255) / 256, 256, 0, stream>>>(msw2, w2pk, 64, 64);
    pack_w_kernel<<<(4096 + 255) / 256, 256, 0, stream>>>(cw1, cw1pk, 64, 64);
    pack_w_kernel<<<(8192 + 255) / 256, 256, 0, stream>>>(nw1, nw1pk, 128, 128);
    pack_w_kernel<<<(4096 + 255) / 256, 256, 0, stream>>>(nw2, nw2pk, 64, 64);

    int eblocks = (E + 63) / 64;
    edge_kernel<<<eblocks, 256, 0, stream>>>(h, x, eidx, eattr, w1pk, msb1, w2pk, msb2,
                                             cw1pk, cb1, cw2, mi_acc, tr_acc, cnt, E);

    int nblocks = (N + 63) / 64;
    node_kernel<<<nblocks, 256, 0, stream>>>(h, x, nw1pk, nb1, nw2pk, nb2, lng, lnb,
                                             cnt, mi_acc, tr_acc, N);
}
